// AHGNN_LP_33346126086888
// MI455X (gfx1250) — matrix operand order repaired and checked by reading, not compiled, not run on hardware
//
#include <hip/hip_runtime.h>
#include <math.h>

typedef _Float16 f16;
typedef __attribute__((ext_vector_type(8)))  _Float16 v8h;
typedef __attribute__((ext_vector_type(16))) _Float16 v16h;
typedef __attribute__((ext_vector_type(8)))  float    v8f;

// ---------------------------------------------------------------------------
// WMMA GEMM: C[M,Nn] = X[M,K] * Y[Nn,K]^T   (f16 inputs, f32 accumulate)
// Block: 256 threads = 8 waves arranged 4(M) x 2(N); block tile 128x64.
// Each wave owns a 32x32 C tile = 2x2 WMMA tiles -> 4 v_wmma per k-step on
// 2 A frags + 2 B frags (2x WMMA-per-byte vs 1x1 tiling, 4 independent
// WMMAs per load-wait window for the XDL pipe).
// Fragment layouts per CDNA5 ISA 7.12.2 (wave32):
//   A 16x32 f16 : lane l<16 -> row M=l, K halves {0..7}+{16..23}; lane>=16 ->
//                 row M=l-16, K halves {8..15}+{24..31}.
//   B 32x16 f16 : lane l<16 -> col N=l, K halves {0..7}+{16..23}; lane>=16 ->
//                 col N=l-16, K halves {8..15}+{24..31} (same as A).
//   C 16x16 f32 : VGPR r -> row m0 + 8*(lane>>4) + r, col n0 + (lane&15).
// ---------------------------------------------------------------------------
__device__ __forceinline__ v16h load_a_frag(const f16* __restrict__ p, int k)
{
    v8h lo = *(const v8h*)(p + k);        // K = k + half*8 + 0..7
    v8h hi = *(const v8h*)(p + k + 16);   // K = k + 16 + half*8 + 0..7
    v16h a;
#pragma unroll
    for (int t = 0; t < 8; ++t) { a[t] = lo[t]; a[8 + t] = hi[t]; }
    return a;
}

__global__ __launch_bounds__(256) void k_gemm_nt_f16f32(
    const f16* __restrict__ X, const f16* __restrict__ Y, float* __restrict__ C,
    int M, int Nn, int K, int ldx, int ldy, int ldc)
{
    const int lane = threadIdx.x & 31;
    const int wave = threadIdx.x >> 5;
    const int wm = wave & 3;        // 0..3 -> M
    const int wn = wave >> 2;       // 0..1 -> N
    const int m0 = blockIdx.y * 128 + wm * 32;
    const int n0 = blockIdx.x * 64 + wn * 32;
    const int half = lane >> 4;     // 0 or 1
    const int l    = lane & 15;

    const f16* __restrict__ A0 = X + (size_t)(m0 + l) * ldx + half * 8;
    const f16* __restrict__ A1 = A0 + (size_t)16 * ldx;
    const f16* __restrict__ B0 = Y + (size_t)(n0 + l) * ldy + half * 8;
    const f16* __restrict__ B1 = B0 + (size_t)16 * ldy;

    v8f c00 = {}, c01 = {}, c10 = {}, c11 = {};
    for (int k = 0; k < K; k += 32) {
        v16h a0 = load_a_frag(A0, k);
        v16h a1 = load_a_frag(A1, k);
        v16h b0 = load_a_frag(B0, k);
        v16h b1 = load_a_frag(B1, k);
        if (k + 32 < K) {
            __builtin_prefetch(A0 + k + 32, 0, 0);
            __builtin_prefetch(A1 + k + 32, 0, 0);
            __builtin_prefetch(B0 + k + 32, 0, 0);
            __builtin_prefetch(B1 + k + 32, 0, 0);
        }
        c00 = __builtin_amdgcn_wmma_f32_16x16x32_f16(false, a0, false, b0, (short)0, c00, false, false);
        c01 = __builtin_amdgcn_wmma_f32_16x16x32_f16(false, a0, false, b1, (short)0, c01, false, false);
        c10 = __builtin_amdgcn_wmma_f32_16x16x32_f16(false, a1, false, b0, (short)0, c10, false, false);
        c11 = __builtin_amdgcn_wmma_f32_16x16x32_f16(false, a1, false, b1, (short)0, c11, false, false);
    }

    float* __restrict__ C00 = C + (size_t)(m0 + half * 8) * ldc + n0 + l;
    float* __restrict__ C10 = C00 + (size_t)16 * ldc;
#pragma unroll
    for (int r = 0; r < 8; ++r) {
        C00[(size_t)r * ldc]      = c00[r];
        C00[(size_t)r * ldc + 16] = c01[r];
        C10[(size_t)r * ldc]      = c10[r];
        C10[(size_t)r * ldc + 16] = c11[r];
    }
}

// ---------------------------------------------------------------------------
// Row L2-normalize -> f16
// ---------------------------------------------------------------------------
__global__ __launch_bounds__(256) void k_normalize_rows(
    const float* __restrict__ X, f16* __restrict__ Zh, int D)
{
    const int i = blockIdx.x;
    const int t = threadIdx.x;
    __shared__ float red[256];
    float acc = 0.f;
    for (int d = t; d < D; d += 256) { float v = X[(size_t)i * D + d]; acc += v * v; }
    red[t] = acc; __syncthreads();
    for (int s = 128; s > 0; s >>= 1) { if (t < s) red[t] += red[t + s]; __syncthreads(); }
    __shared__ float inv;
    if (t == 0) { float n = sqrtf(red[0]); inv = 1.0f / fmaxf(n, 1e-12f); }
    __syncthreads();
    for (int d = t; d < D; d += 256)
        Zh[(size_t)i * D + d] = (f16)(X[(size_t)i * D + d] * inv);
}

// per-node halves of the edge score: s1[i]=emb[i].W[0:D], s2[i]=emb[i].W[D:2D]
__global__ __launch_bounds__(256) void k_node_scores(
    const float* __restrict__ X, const float* __restrict__ W,
    float* __restrict__ s1, float* __restrict__ s2, int D)
{
    const int i = blockIdx.x;
    const int t = threadIdx.x;
    __shared__ float r1[256], r2[256];
    float a = 0.f, b = 0.f;
    for (int d = t; d < D; d += 256) {
        float e = X[(size_t)i * D + d];
        a += e * W[d];
        b += e * W[D + d];
    }
    r1[t] = a; r2[t] = b; __syncthreads();
    for (int s = 128; s > 0; s >>= 1) {
        if (t < s) { r1[t] += r1[t + s]; r2[t] += r2[t + s]; }
        __syncthreads();
    }
    if (t == 0) { s1[i] = r1[0]; s2[i] = r2[0]; }
}

// transpose embedding: embT_h[d*N + n] = (f16) emb[n*D + d]
__global__ __launch_bounds__(256) void k_transpose_emb(
    const float* __restrict__ X, f16* __restrict__ XT, int N, int D)
{
    size_t t = (size_t)blockIdx.x * blockDim.x + threadIdx.x;
    size_t total = (size_t)N * D;
    if (t >= total) return;
    int d = (int)(t / N), n = (int)(t % N);
    XT[t] = (f16)X[(size_t)n * D + d];
}

__global__ void k_init_seg(unsigned* __restrict__ smax, float* __restrict__ ssum, int N)
{
    int i = blockIdx.x * blockDim.x + threadIdx.x;
    if (i < N) { smax[i] = 0u; ssum[i] = 0.f; }
}

__device__ __forceinline__ unsigned fmap(float x) {
    unsigned u = __float_as_uint(x);
    return (u >> 31) ? ~u : (u | 0x80000000u);
}
__device__ __forceinline__ float funmap(unsigned m) {
    return (m & 0x80000000u) ? __uint_as_float(m & 0x7FFFFFFFu) : __uint_as_float(~m);
}

// edge pass A: leaky-relu score + segment max (monotone-uint atomicMax)
__global__ __launch_bounds__(256) void k_edge_a(
    const int* __restrict__ src, const int* __restrict__ dst,
    const float* __restrict__ s1, const float* __restrict__ s2,
    const float* __restrict__ bsc, float* __restrict__ xs,
    unsigned* __restrict__ smax, int E)
{
    int e = blockIdx.x * blockDim.x + threadIdx.x;
    if (e >= E) return;
    float x = s1[src[e]] + s2[dst[e]] + bsc[0];
    x = (x > 0.f) ? x : 0.01f * x;            // leaky_relu, slope 0.01
    xs[e] = x;
    atomicMax(&smax[src[e]], fmap(x));
}

// edge pass B: exp(x - max) + segment sum
__global__ __launch_bounds__(256) void k_edge_b(
    const int* __restrict__ src, const float* __restrict__ xs,
    const unsigned* __restrict__ smax, float* __restrict__ eexp,
    float* __restrict__ ssum, int E)
{
    int e = blockIdx.x * blockDim.x + threadIdx.x;
    if (e >= E) return;
    float m = funmap(smax[src[e]]);
    float v = __expf(xs[e] - m);
    eexp[e] = v;
    atomicAdd(&ssum[src[e]], v);
}

// edge pass C: normalized score scattered into fitness (already = connectivity)
__global__ __launch_bounds__(256) void k_edge_c(
    const int* __restrict__ src, const int* __restrict__ dst,
    const float* __restrict__ eexp, const float* __restrict__ ssum,
    float* __restrict__ fitness, int E, int N)
{
    int e = blockIdx.x * blockDim.x + threadIdx.x;
    if (e >= E) return;
    int s = src[e];
    float sc = eexp[e] / (ssum[s] + 1e-16f);
    atomicAdd(&fitness[(size_t)s * N + dst[e]], sc);
}

// A1 = offdiag(edge_matrix) kept where fitness >= 0; also emit A1^T (both f16)
__global__ __launch_bounds__(256) void k_build_A1(
    const float* __restrict__ EM, const float* __restrict__ F,
    f16* __restrict__ A1, f16* __restrict__ A1T, int N)
{
    size_t t = (size_t)blockIdx.x * blockDim.x + threadIdx.x;
    size_t total = (size_t)N * N;
    if (t >= total) return;
    int i = (int)(t / N), j = (int)(t % N);
    float v = (i != j && EM[t] != 0.0f && F[t] >= 0.0f) ? 1.0f : 0.0f;
    A1[t] = (f16)v;
    A1T[(size_t)j * N + i] = (f16)v;
}

// cluster_scores: mean fitness over A1 and A2 neighborhoods (A2 on the fly)
__global__ __launch_bounds__(256) void k_scores(
    const f16* __restrict__ A1, const float* __restrict__ P,
    const float* __restrict__ F, float* __restrict__ cs, int N)
{
    const int i = blockIdx.x;
    const int t = threadIdx.x;
    __shared__ float r0[256], r1[256], r2[256], r3[256];
    float d1 = 0.f, f1 = 0.f, d2 = 0.f, f2 = 0.f;
    for (int j = t; j < N; j += 256) {
        size_t idx = (size_t)i * N + j;
        float a1 = (float)A1[idx];
        float fit = F[idx];
        d1 += a1; f1 += fit * a1;
        float a2 = (j != i && a1 == 0.0f && P[idx] > 0.5f) ? 1.0f : 0.0f;
        d2 += a2; f2 += fit * a2;
    }
    r0[t] = d1; r1[t] = f1; r2[t] = d2; r3[t] = f2; __syncthreads();
    for (int s = 128; s > 0; s >>= 1) {
        if (t < s) { r0[t]+=r0[t+s]; r1[t]+=r1[t+s]; r2[t]+=r2[t+s]; r3[t]+=r3[t+s]; }
        __syncthreads();
    }
    if (t == 0) {
        float sc = (r0[0] > 0.f ? r1[0] / r0[0] : 0.f)
                 + (r2[0] > 0.f ? r3[0] / r2[0] : 0.f);
        cs[i] = sc * 0.5f;   // / CLUSTER_RANGE
    }
}

// is_cluster[i] = all_j (cs[i] - A1[i,j]*cs[j] > 0)
__global__ __launch_bounds__(256) void k_iscluster(
    const f16* __restrict__ A1, const float* __restrict__ cs,
    int* __restrict__ iscl, int N)
{
    const int i = blockIdx.x;
    const int t = threadIdx.x;
    __shared__ int r[256];
    float ci = cs[i];
    int ok = 1;
    for (int j = t; j < N; j += 256) {
        float a1 = (float)A1[(size_t)i * N + j];
        ok &= (ci - a1 * cs[j] > 0.0f) ? 1 : 0;
    }
    r[t] = ok; __syncthreads();
    for (int s = 128; s > 0; s >>= 1) { if (t < s) r[t] &= r[t + s]; __syncthreads(); }
    if (t == 0) iscl[i] = r[0];
}

// rowflag[i] = any_j (S[i,j] && is_cluster[j])   (S = cluster_matrix, on the fly)
__global__ __launch_bounds__(256) void k_rowflag(
    const f16* __restrict__ A1, const float* __restrict__ P,
    const int* __restrict__ iscl, int* __restrict__ rowflag, int N)
{
    const int i = blockIdx.x;
    const int t = threadIdx.x;
    __shared__ int r[256];
    int fl = 0;
    for (int j = t; j < N; j += 256) {
        size_t idx = (size_t)i * N + j;
        int s = (i != j) && ((float)A1[idx] != 0.0f || P[idx] > 0.5f);
        fl |= (s && iscl[j]);
    }
    r[t] = fl; __syncthreads();
    for (int s = 128; s > 0; s >>= 1) { if (t < s) r[t] |= r[t + s]; __syncthreads(); }
    if (t == 0) rowflag[i] = r[0];
}

// column sums of S -> isolated; combine into reduced / keeping / col_keep
__global__ __launch_bounds__(256) void k_colreduce(
    const f16* __restrict__ A1, const float* __restrict__ P,
    const int* __restrict__ rowflag, const int* __restrict__ iscl,
    int* __restrict__ keeping, int* __restrict__ colkeep, int N)
{
    int j = blockIdx.x * blockDim.x + threadIdx.x;
    if (j >= N) return;
    float colsum = 0.f;
    for (int i = 0; i < N; ++i) {
        size_t idx = (size_t)i * N + j;   // coalesced across threads (fixed i)
        int s = (i != j) && ((float)A1[idx] != 0.0f || P[idx] > 0.5f);
        colsum += (float)s;
    }
    int isolated = (colsum == 0.f);
    int reduced  = rowflag[j] | isolated;
    keeping[j] = (!iscl[j]) && (!reduced);
    colkeep[j] = !reduced;
}

// S_f^T in f16: SfT[j,i] = colkeep[j] * (i==j ? 1 : (keeping[j] ? 0 : S*fitness))
__global__ __launch_bounds__(256) void k_build_SfT(
    const f16* __restrict__ A1, const float* __restrict__ P,
    const float* __restrict__ F, const int* __restrict__ keeping,
    const int* __restrict__ colkeep, f16* __restrict__ SfT, int N)
{
    size_t t = (size_t)blockIdx.x * blockDim.x + threadIdx.x;
    size_t total = (size_t)N * N;
    if (t >= total) return;
    int j = (int)(t / N), i = (int)(t % N);
    size_t idx = (size_t)i * N + j;
    int s = (i != j) && ((float)A1[idx] != 0.0f || P[idx] > 0.5f);
    float v;
    if (i == j)             v = 1.0f;
    else if (keeping[j])    v = 0.0f;
    else                    v = s ? F[idx] : 0.0f;
    if (!colkeep[j])        v = 0.0f;
    SfT[t] = (f16)v;
}

// ---------------------------------------------------------------------------
extern "C" void kernel_launch(void* const* d_in, const int* in_sizes, int n_in,
                              void* d_out, int out_size, void* d_ws, size_t ws_size,
                              hipStream_t stream)
{
    const float* emb = (const float*)d_in[0];
    const int*   ei  = (const int*)d_in[1];
    const float* EM  = (const float*)d_in[2];
    // d_in[3] = edge_matrix_weight (== edge_matrix here, unused by reference math)
    const float* W   = (const float*)d_in[4];
    const float* bsc = (const float*)d_in[5];
    float* out = (float*)d_out;

    const int E = in_sizes[1] / 2;
    const int N = (int)(sqrt((double)in_sizes[2]) + 0.5);
    const int D = in_sizes[0] / N;
    const int* src = ei;
    const int* dst = ei + E;

    // workspace carve-up (256B aligned)
    char* p = (char*)d_ws;
    auto alloc = [&](size_t bytes) -> void* {
        void* r = (void*)p; p += (bytes + 255) & ~(size_t)255; return r;
    };
    f16*      Zh     = (f16*)alloc((size_t)N * D * sizeof(f16));
    f16*      embTh  = (f16*)alloc((size_t)N * D * sizeof(f16));
    float*    fit    = (float*)alloc((size_t)N * N * sizeof(float));
    f16*      A1h    = (f16*)alloc((size_t)N * N * sizeof(f16));
    f16*      A1Th   = (f16*)alloc((size_t)N * N * sizeof(f16));
    float*    A1A1   = (float*)alloc((size_t)N * N * sizeof(float));
    f16*      SfTh   = (f16*)alloc((size_t)N * N * sizeof(f16));
    float*    s1     = (float*)alloc((size_t)N * sizeof(float));
    float*    s2     = (float*)alloc((size_t)N * sizeof(float));
    unsigned* smax   = (unsigned*)alloc((size_t)N * sizeof(unsigned));
    float*    ssum   = (float*)alloc((size_t)N * sizeof(float));
    float*    xs     = (float*)alloc((size_t)E * sizeof(float));
    float*    eexp   = (float*)alloc((size_t)E * sizeof(float));
    float*    cs     = (float*)alloc((size_t)N * sizeof(float));
    int*      iscl   = (int*)alloc((size_t)N * sizeof(int));
    int*      rowfl  = (int*)alloc((size_t)N * sizeof(int));
    int*      keepg  = (int*)alloc((size_t)N * sizeof(int));
    int*      ckeep  = (int*)alloc((size_t)N * sizeof(int));
    (void)ws_size; (void)n_in; (void)out_size;

    const int eBlocks  = (E + 255) / 256;
    const int nBlocks  = (N + 255) / 256;
    const int nnBlocks = (int)(((size_t)N * N + 255) / 256);
    const int ndBlocks = (int)(((size_t)N * D + 255) / 256);

    // 1) Z (f16), per-node score halves, embedding transpose, seg init
    k_normalize_rows<<<N, 256, 0, stream>>>(emb, Zh, D);
    k_node_scores  <<<N, 256, 0, stream>>>(emb, W, s1, s2, D);
    k_transpose_emb<<<ndBlocks, 256, 0, stream>>>(emb, embTh, N, D);
    k_init_seg     <<<nBlocks, 256, 0, stream>>>(smax, ssum, N);

    // 2) connectivity: fitness = Zh . Zh^T    (WMMA, 128x64 block tile)
    {
        dim3 g(N / 64, N / 128);
        k_gemm_nt_f16f32<<<g, 256, 0, stream>>>(Zh, Zh, fit, N, N, D, D, D, N);
    }

    // 3) structural fitness: segment softmax over src, scatter-add into fitness
    k_edge_a<<<eBlocks, 256, 0, stream>>>(src, dst, s1, s2, bsc, xs, smax, E);
    k_edge_b<<<eBlocks, 256, 0, stream>>>(src, xs, smax, eexp, ssum, E);
    k_edge_c<<<eBlocks, 256, 0, stream>>>(src, dst, eexp, ssum, fit, E, N);

    // 4) A1 (+ transpose), then A1@A1 via WMMA (exact: 0/1 values, f32 accum)
    k_build_A1<<<nnBlocks, 256, 0, stream>>>(EM, fit, A1h, A1Th, N);
    {
        dim3 g(N / 64, N / 128);
        k_gemm_nt_f16f32<<<g, 256, 0, stream>>>(A1h, A1Th, A1A1, N, N, N, N, N, N);
    }

    // 5) cluster scores, extrema selection, reduction masks
    k_scores    <<<N, 256, 0, stream>>>(A1h, A1A1, fit, cs, N);
    k_iscluster <<<N, 256, 0, stream>>>(A1h, cs, iscl, N);
    k_rowflag   <<<N, 256, 0, stream>>>(A1h, A1A1, iscl, rowfl, N);
    k_colreduce <<<nBlocks, 256, 0, stream>>>(A1h, A1A1, rowfl, iscl, keepg, ckeep, N);

    // 6) assignment matrix (transposed, f16) and pooled = SfT . emb  (WMMA)
    k_build_SfT<<<nnBlocks, 256, 0, stream>>>(A1h, A1A1, fit, keepg, ckeep, SfTh, N);
    {
        dim3 g(D / 64, N / 128);
        k_gemm_nt_f16f32<<<g, 256, 0, stream>>>(SfTh, embTh, out, N, D, N, N, N, D);
    }
}
